// SE3PointKernel_44787918963412
// MI455X (gfx1250) — matrix operand order repaired and checked by reading, not compiled, not run on hardware
//
#include <hip/hip_runtime.h>
#include <hip/hip_bf16.h>

// SE(3) point-conv kernel generator, fused for MI455X (gfx1250, wave32, WMMA).
//   phase A: h = relu(r*W1+b1)  (K=64)  -> f16 A-operand tiles
//   phase B: weights = h @ W2 + b2 (N=384) via v_wmma_f32_16x16x32_f16
//   phase C: CG contraction -> 32x32 block per point pair, streamed with NT stores.

typedef __attribute__((ext_vector_type(16))) _Float16 v16h;
typedef __attribute__((ext_vector_type(8)))  float    v8f;

#define NHID       64
#define NPATH      384
#define NPTS       131072        // B * N_OUT_PTS * N_IN_PTS = 2*256*256
#define WG_THREADS 128           // 4 waves, each wave owns 32 points
#define WSTRIDE    193           // padded per-point weight row (dwords); 193%64==1 -> conflict-free

__global__ __launch_bounds__(WG_THREADS)
void se3_kernel(const float* __restrict__ dm,
                const float* __restrict__ W1, const float* __restrict__ b1,
                const float* __restrict__ W2, const float* __restrict__ b2,
                const float* __restrict__ Q00, const float* __restrict__ Q01,
                const float* __restrict__ Q10, const float* __restrict__ Q11,
                float* __restrict__ out)
{
    extern __shared__ char smem_raw[];
    float*     sW1 = (float*)smem_raw;          // 64 f32
    float*     sB1 = sW1 + 64;                  // 64 f32
    float*     sB2 = sB1 + 64;                  // 384 f32
    _Float16*  sW2 = (_Float16*)(sB2 + 384);    // 64*384 halves, WMMA-B swizzled (48KB)
    float*     sWt = (float*)(sW2 + NHID * NPATH); // 4 waves * 32 pts * WSTRIDE f32

    const int tid  = threadIdx.x;
    const int lane = tid & 31;
    const int wave = tid >> 5;

    // ---- stage constants into LDS -------------------------------------------
    for (int i = tid; i < 64; i += WG_THREADS) { sW1[i] = W1[i]; sB1[i] = b1[i]; }
    for (int i = tid; i < NPATH; i += WG_THREADS) sB2[i] = b2[i];
    // W2 (f32, row-major 64x384) -> f16 in WMMA B-operand order:
    //   B tile (k-chunk c, n-tile nt): lane l holds column n = 16*nt + (l&15),
    //   halves 0..15 = k = 32*c + 16*(l>>4) + h
    for (int idx = tid; idx < NHID * NPATH; idx += WG_THREADS) {
        int k = idx / NPATH, n = idx % NPATH;
        int nt = n >> 4, c = k >> 5;
        int laneSlot = (n & 15) + (((k >> 4) & 1) << 4);
        int half     = k & 15;
        sW2[(((nt * 2 + c) * 32) + laneSlot) * 16 + half] = (_Float16)W2[idx];
    }
    __syncthreads();

    // ---- per-lane point geometry --------------------------------------------
    const int pt = blockIdx.x * WG_THREADS + tid;    // linear (b,n,m); m fastest
    float x = dm[pt * 3 + 0], y = dm[pt * 3 + 1], z = dm[pt * 3 + 2];
    float r = sqrtf(x * x + y * y + z * z);
    float inv = 1.0f / fmaxf(r, 1e-12f);
    float ux = x * inv, uy = y * inv, uz = z * inv;

    const float C0  = 0.28209479177387814f;
    const float C1  = 0.4886025119029199f;
    const float C2A = 1.0925484305920792f;
    const float C2B = 0.31539156525252005f;
    const float C2C = 0.5462742152960396f;
    float Y1[3] = { C1 * uy, C1 * uz, C1 * ux };
    float Y2[5] = { C2A * ux * uy, C2A * uy * uz, C2B * (3.0f * uz * uz - 1.0f),
                    C2A * ux * uz, C2C * (ux * ux - uy * uy) };

    // normalization: sqrt(2*l_in+1)*sqrt(4pi)/sqrt(num); num = {4096, 8192}
    const float S4PI  = 3.5449077018110318f;
    const float SQRT3 = 1.7320508075688772f;
    const float f00 = S4PI / 64.0f;
    const float f01 = SQRT3 * S4PI / 64.0f;
    const float f10 = S4PI / 90.50966799187809f;
    const float f11 = SQRT3 * S4PI / 90.50966799187809f;

    // fold CG tensors with Y and norms (per lane)
    float A0f = f00 * Q00[0] * C0;
    float Gf[3], Hf[3];
#pragma unroll
    for (int j = 0; j < 3; ++j) {
        Gf[j] = f01 * (Q01[j * 3 + 0] * Y1[0] + Q01[j * 3 + 1] * Y1[1] + Q01[j * 3 + 2] * Y1[2]);
        Hf[j] = f10 * (Q10[j * 3 + 0] * Y1[0] + Q10[j * 3 + 1] * Y1[1] + Q10[j * 3 + 2] * Y1[2]);
    }
    float T0f[9], T1f[9], T2f[9];
#pragma unroll
    for (int ij = 0; ij < 9; ++ij) {
        const float* q = Q11 + ij * 9;
        T0f[ij] = f11 * q[0] * C0;
        T1f[ij] = f11 * (q[1] * Y1[0] + q[2] * Y1[1] + q[3] * Y1[2]);
        T2f[ij] = f11 * (q[4] * Y2[0] + q[5] * Y2[1] + q[6] * Y2[2] +
                         q[7] * Y2[3] + q[8] * Y2[4]);
    }

    // ---- build A-operand tiles: h = relu(r*W1+b1) in f16, contiguous-k A layout ------
    // 16-bit A 16x32: lanes 0-15 row M=lane,  halves = K{0..15}+32c
    //                 lanes16-31 row M=l-16,  halves = K{16..31}+32c
    v16h A[2][2];
#pragma unroll
    for (int t = 0; t < 2; ++t) {
        float rr = __shfl(r, t * 16 + (lane & 15), 32);
        int kb = (lane >> 4) << 4;
#pragma unroll
        for (int c = 0; c < 2; ++c) {
            v16h a;
#pragma unroll
            for (int i = 0; i < 16; ++i) {
                int k = c * 32 + kb + i;
                float hv = fmaxf(rr * sW1[k] + sB1[k], 0.0f);
                a[i] = (_Float16)hv;
            }
            A[t][c] = a;
        }
    }

    float* wreg = sWt + (size_t)(wave * 32) * WSTRIDE;        // wave-private staging
    float* mw   = sWt + (size_t)(wave * 32 + lane) * WSTRIDE; // this lane's point row
    float* outp = out + pt;                                   // + (row*32+col)*NPTS

    // weights[n0 .. n0+16*ntiles) for this wave's 32 points -> LDS (point-major)
    auto compute_weights = [&](int n0, int ntiles) {
        const int pl    = lane & 15;
        const int pbase = (lane >> 4) << 3;
        for (int ntl = 0; ntl < ntiles; ++ntl) {
            int ntg = (n0 >> 4) + ntl;
            v16h b0 = *(const v16h*)(sW2 + (size_t)((ntg * 2 + 0) * 32 + lane) * 16);
            v16h b1v = *(const v16h*)(sW2 + (size_t)((ntg * 2 + 1) * 32 + lane) * 16);
            float bias = sB2[n0 + ntl * 16 + pl];
#pragma unroll
            for (int t = 0; t < 2; ++t) {
                v8f acc = {0.f, 0.f, 0.f, 0.f, 0.f, 0.f, 0.f, 0.f};
                acc = __builtin_amdgcn_wmma_f32_16x16x32_f16(
                          false, A[t][0], false, b0,  (short)0, acc, false, false);
                acc = __builtin_amdgcn_wmma_f32_16x16x32_f16(
                          false, A[t][1], false, b1v, (short)0, acc, false, false);
                // C layout: acc[g], lane l -> point M = g + 8*(l>>4), path N = l&15
#pragma unroll
                for (int g = 0; g < 8; ++g) {
                    int p = t * 16 + pbase + g;
                    wreg[p * WSTRIDE + ntl * 16 + pl] = acc[g] + bias;
                }
            }
        }
        asm volatile("s_wait_dscnt 0" ::: "memory");
    };

    // ---- block (out l=0, in l=0): paths 0..63, rows 0..7, cols 0..7 ---------
    compute_weights(0, 4);
    for (int u = 0; u < 8; ++u)
        for (int v = 0; v < 8; ++v) {
            float w = mw[u * 8 + v];
            __builtin_nontemporal_store(A0f * w, outp + (size_t)(u * 32 + v) * NPTS);
        }
    asm volatile("" ::: "memory");

    // ---- block (out l=0, in l=1): paths 64..127, rows 0..7, cols 8..31 ------
    compute_weights(64, 4);
    for (int u = 0; u < 8; ++u)
        for (int v = 0; v < 8; ++v) {
            float w = mw[u * 8 + v];
#pragma unroll
            for (int j = 0; j < 3; ++j)
                __builtin_nontemporal_store(Gf[j] * w,
                    outp + (size_t)(u * 32 + 8 + v * 3 + j) * NPTS);
        }
    asm volatile("" ::: "memory");

    // ---- block (out l=1, in l=0): paths 128..191, rows 8..31, cols 0..7 -----
    compute_weights(128, 4);
    for (int u = 0; u < 8; ++u)
        for (int v = 0; v < 8; ++v) {
            float w = mw[u * 8 + v];
#pragma unroll
            for (int i = 0; i < 3; ++i)
                __builtin_nontemporal_store(Hf[i] * w,
                    outp + (size_t)((8 + u * 3 + i) * 32 + v) * NPTS);
        }
    asm volatile("" ::: "memory");

    // ---- block (out l=1, in l=1): paths 192..383, rows 8..31, cols 8..31 ----
    compute_weights(192, 12);
    for (int u = 0; u < 8; ++u)
        for (int v = 0; v < 8; ++v) {
            int base = (u * 8 + v) * 3;
            float w0 = mw[base + 0], w1 = mw[base + 1], w2 = mw[base + 2];
#pragma unroll
            for (int i = 0; i < 3; ++i)
#pragma unroll
                for (int j = 0; j < 3; ++j) {
                    float val = T0f[i * 3 + j] * w0 + T1f[i * 3 + j] * w1 + T2f[i * 3 + j] * w2;
                    __builtin_nontemporal_store(val,
                        outp + (size_t)((8 + u * 3 + i) * 32 + 8 + v * 3 + j) * NPTS);
                }
        }
}

extern "C" void kernel_launch(void* const* d_in, const int* in_sizes, int n_in,
                              void* d_out, int out_size, void* d_ws, size_t ws_size,
                              hipStream_t stream) {
    const float* dm  = (const float*)d_in[0];
    const float* W1  = (const float*)d_in[1];
    const float* b1  = (const float*)d_in[2];
    const float* W2  = (const float*)d_in[3];
    const float* b2  = (const float*)d_in[4];
    const float* Q00 = (const float*)d_in[5];
    const float* Q01 = (const float*)d_in[6];
    const float* Q10 = (const float*)d_in[7];
    const float* Q11 = (const float*)d_in[8];
    float* out = (float*)d_out;

    size_t smem = (size_t)(64 + 64 + NPATH) * 4       // W1,b1,b2
                + (size_t)NHID * NPATH * 2            // W2 as f16 (48KB)
                + (size_t)4 * 32 * WSTRIDE * 4;       // per-wave weight staging
    dim3 grid(NPTS / WG_THREADS), block(WG_THREADS);
    hipLaunchKernelGGL(se3_kernel, grid, block, smem, stream,
                       dm, W1, b1, W2, b2, Q00, Q01, Q10, Q11, out);
}
